// ETCContext_48490180772018
// MI455X (gfx1250) — compile-verified, matrix operand order checked by reading, not run on hardware
//
#include <hip/hip_runtime.h>

#define HB   8
#define LL   1024
#define DD   512
#define DH   64
#define TL   16
#define TM   32
#define NCH  (LL / TM)   // 32 m-chunks

// LDS byte offsets (dynamic smem starts at LDS offset 0; no static LDS in kernel)
#define KS_OFF   0u
#define QS_OFF   16384u
#define VS_OFF   49152u
#define SBUF_OFF 81920u
#define PS_OFF   98304u
#define COL_OFF  106496u
#define SMEM_BYTES 110592

typedef __attribute__((ext_vector_type(16))) __bf16 v16bf;
typedef __attribute__((ext_vector_type(8)))  float  v8f;
typedef __attribute__((ext_vector_type(4)))  unsigned int u32x4;
typedef __attribute__((ext_vector_type(8)))  unsigned int u32x8;

__device__ __forceinline__ unsigned short f2bf(float f) {
  unsigned int x = __float_as_uint(f);
  unsigned int r = (x + 0x7FFFu + ((x >> 16) & 1u)) >> 16;  // round-to-nearest-even
  return (unsigned short)r;
}
__device__ __forceinline__ __bf16 us2bf(unsigned short u) {
  union { unsigned short u; __bf16 b; } c; c.u = u; return c.b;
}
// K-index for slot r, element e of a 16-bit WMMA A/B operand (16x32 / 32x16).
__device__ __forceinline__ int kidx(int r, int e, int kadd) {
  return ((r & 3) << 1) + e + ((r >> 2) << 4) + kadd;
}

// --- Tensor Data Mover: DMA a [rows x 512] bf16 tile (row-major, contiguous)
// from global memory into LDS at byte offset ldsOff. Wave-uniform; EXEC ignored.
__device__ __forceinline__ void tdm_load_2d(unsigned int ldsOff,
                                            const unsigned short* gptr,
                                            unsigned int rows) {
  unsigned long long ga = (unsigned long long)(uintptr_t)gptr;
  u32x4 g0;
  g0[0] = 1u;                                            // count=1 (valid), user mode
  g0[1] = ldsOff;                                        // lds_addr (bytes)
  g0[2] = (unsigned int)(ga & 0xFFFFFFFFull);            // global_addr[31:0]
  g0[3] = (unsigned int)((ga >> 32) & 0x01FFFFFFull)     // global_addr[56:32]
        | (2u << 30);                                    // type=2 ("image")
  u32x8 g1;
  g1[0] = (1u << 16);          // workgroup_mask=0, data_size=1 (2 bytes)
  g1[1] = (DD & 0xFFFFu) << 16;            // tensor_dim0[15:0] = 512
  g1[2] = (DD >> 16) | (rows << 16);       // tensor_dim0[31:16]=0, tensor_dim1[15:0]=rows
  g1[3] = ((unsigned)DD << 16);            // tensor_dim1 hi=0, tile_dim0 = 512
  g1[4] = rows;                            // tile_dim1 = rows, tile_dim2 = 0
  g1[5] = (unsigned)DD;                    // tensor_dim0_stride[31:0] = 512
  g1[6] = 0u;                              // stride hi, tensor_dim1_stride lo
  g1[7] = 0u;
  asm volatile("tensor_load_to_lds %0, %1" :: "s"(g0), "s"(g1) : "memory");
}

// f32 -> packed bf16 conversion pre-pass (so TDM can move bf16 directly)
__global__ __launch_bounds__(256)
void to_bf16_kernel(const float* __restrict__ src, unsigned short* __restrict__ dst) {
  int i = (blockIdx.x * 256 + threadIdx.x) * 4;
  float4 f = *reinterpret_cast<const float4*>(src + i);
  dst[i + 0] = f2bf(f.x); dst[i + 1] = f2bf(f.y);
  dst[i + 2] = f2bf(f.z); dst[i + 3] = f2bf(f.w);
}

__global__ __launch_bounds__(256, 2)
void etc_attn_kernel(const unsigned short* __restrict__ Kb,
                     const unsigned short* __restrict__ Qb,
                     const unsigned short* __restrict__ Vb,
                     const float* __restrict__ doc,
                     const float* __restrict__ gamma, const float* __restrict__ beta,
                     float* __restrict__ out, float* __restrict__ wAcc)
{
  extern __shared__ char smem[];
  unsigned short* kS  = (unsigned short*)(smem + KS_OFF);   // [16][512] bf16
  unsigned short* qS  = (unsigned short*)(smem + QS_OFF);   // [32][512] bf16
  unsigned short* vS  = (unsigned short*)(smem + VS_OFF);   // [32][512] bf16
  float*          sBuf= (float*)(smem + SBUF_OFF);          // [8][16][32] f32
  unsigned short* pS  = (unsigned short*)(smem + PS_OFF);   // [8][16][32] bf16
  float*          colFull = (float*)(smem + COL_OFF);       // [1024] f32

  const int tid     = threadIdx.x;
  const int lane    = tid & 31;
  const int h       = tid >> 5;          // wave id == head id
  const int b       = blockIdx.x >> 6;
  const int l0      = (blockIdx.x & 63) * TL;
  const int lenB    = (int)doc[b];
  const int mcol    = lane & 15;
  const int halfOff = (lane & 16) ? 8 : 0;
  const int kadd    = (lane & 16) ? 8 : 0;

  float g[HB], be[HB];
  #pragma unroll
  for (int i = 0; i < HB; ++i) { g[i] = gamma[i]; be[i] = beta[i]; }

  // Stage K l-tile once via TDM; zero local column-sum accumulator.
  if (h == 0) {
    tdm_load_2d(KS_OFF, Kb + ((size_t)b * LL + l0) * DD, TL);
    __builtin_amdgcn_s_wait_tensorcnt(0);
  }
  for (int i = tid; i < LL; i += 256) colFull[i] = 0.0f;
  __syncthreads();

  // Persistent A operands: K tile of this head, two d-chunks of 32.
  v16bf aK0, aK1;
  #pragma unroll
  for (int r = 0; r < 8; ++r)
    #pragma unroll
    for (int e = 0; e < 2; ++e) {
      int k = kidx(r, e, kadd);
      aK0[2 * r + e] = us2bf(kS[mcol * DD + h * DH +  0 + k]);
      aK1[2 * r + e] = us2bf(kS[mcol * DD + h * DH + 32 + k]);
    }

  // S-tile compute (4 WMMAs) + scatter to sBuf in C-layout order.
  auto computeS = [&]() {
    v8f s0 = {}, s1 = {};
    #pragma unroll
    for (int c = 0; c < 2; ++c) {
      v16bf bq0, bq1;
      #pragma unroll
      for (int r = 0; r < 8; ++r)
        #pragma unroll
        for (int e = 0; e < 2; ++e) {
          int k = kidx(r, e, kadd);
          bq0[2 * r + e] = us2bf(qS[(     mcol) * DD + h * DH + c * 32 + k]);
          bq1[2 * r + e] = us2bf(qS[(16 + mcol) * DD + h * DH + c * 32 + k]);
        }
      v16bf a = c ? aK1 : aK0;
      s0 = __builtin_amdgcn_wmma_f32_16x16x32_bf16(false, a, false, bq0, (short)0, s0, false, false);
      s1 = __builtin_amdgcn_wmma_f32_16x16x32_bf16(false, a, false, bq1, (short)0, s1, false, false);
    }
    #pragma unroll
    for (int r = 0; r < 8; ++r) {
      sBuf[h * 512 + (r + halfOff) * 32 +      mcol] = s0[r];
      sBuf[h * 512 + (r + halfOff) * 32 + 16 + mcol] = s1[r];
    }
  };

  // LayerNorm over the 8 heads at each (l,m), then pad-mask to -1e30.
  auto layernormMask = [&](int m0) {
    for (int p = tid; p < TL * TM; p += 256) {
      int l = p >> 5, m = p & 31;
      float vals[HB], mu = 0.0f;
      #pragma unroll
      for (int hh = 0; hh < HB; ++hh) { vals[hh] = sBuf[hh * 512 + l * 32 + m]; mu += vals[hh]; }
      mu *= 0.125f;
      float var = 0.0f;
      #pragma unroll
      for (int hh = 0; hh < HB; ++hh) { float dlt = vals[hh] - mu; var += dlt * dlt; }
      var *= 0.125f;
      float rs = rsqrtf(var + 1e-5f);
      bool ok = (l0 + l < lenB) && (m0 + m < lenB);
      #pragma unroll
      for (int hh = 0; hh < HB; ++hh)
        sBuf[hh * 512 + l * 32 + m] = ok ? ((vals[hh] - mu) * rs * g[hh] + be[hh]) : -1e30f;
    }
  };

  float Mrow[8], Srow[8], invS[8];
  #pragma unroll
  for (int r = 0; r < 8; ++r) { Mrow[r] = -1e30f; Srow[r] = 0.0f; }

  // ---------------- Pass A: online softmax row statistics ----------------
  for (int ch = 0; ch < NCH; ++ch) {
    const int m0 = ch * TM;
    if (h == 0) {
      tdm_load_2d(QS_OFF, Qb + ((size_t)b * LL + m0) * DD, TM);
      __builtin_amdgcn_s_wait_tensorcnt(0);
    }
    if (ch + 1 < NCH)
      __builtin_prefetch(Qb + ((size_t)b * LL + m0 + TM) * DD + tid * 64, 0, 1);
    __syncthreads();
    computeS();
    __syncthreads();
    layernormMask(m0);
    __syncthreads();
    #pragma unroll
    for (int r = 0; r < 8; ++r) {
      float v0 = sBuf[h * 512 + (r + halfOff) * 32 +      mcol];
      float v1 = sBuf[h * 512 + (r + halfOff) * 32 + 16 + mcol];
      float tm = fmaxf(v0, v1);
      #pragma unroll
      for (int msk = 1; msk < 16; msk <<= 1) tm = fmaxf(tm, __shfl_xor(tm, msk, 32));
      float nM = fmaxf(Mrow[r], tm);
      float ts = __expf(v0 - nM) + __expf(v1 - nM);
      #pragma unroll
      for (int msk = 1; msk < 16; msk <<= 1) ts += __shfl_xor(ts, msk, 32);
      Srow[r] = Srow[r] * __expf(Mrow[r] - nM) + ts;
      Mrow[r] = nM;
    }
  }

  #pragma unroll
  for (int r = 0; r < 8; ++r) {
    bool alive = (l0 + r + halfOff) < lenB;       // fully-masked row -> softmax NaN -> 0
    invS[r] = alive ? (1.0f / Srow[r]) : 0.0f;
  }

  // ---------------- Pass B: normalized P, column sums, O = P @ V ----------------
  v8f O0 = {}, O1 = {}, O2 = {}, O3 = {};
  for (int ch = 0; ch < NCH; ++ch) {
    const int m0 = ch * TM;
    if (h == 0) {
      tdm_load_2d(QS_OFF, Qb + ((size_t)b * LL + m0) * DD, TM);
      tdm_load_2d(VS_OFF, Vb + ((size_t)b * LL + m0) * DD, TM);
      __builtin_amdgcn_s_wait_tensorcnt(0);
    }
    __syncthreads();
    computeS();
    __syncthreads();
    layernormMask(m0);
    __syncthreads();

    // exact normalized probabilities (C-layout) + per-column partial sums
    float c0 = 0.0f, c1 = 0.0f;
    #pragma unroll
    for (int r = 0; r < 8; ++r) {
      float v0 = sBuf[h * 512 + (r + halfOff) * 32 +      mcol];
      float v1 = sBuf[h * 512 + (r + halfOff) * 32 + 16 + mcol];
      float p0 = __expf(v0 - Mrow[r]) * invS[r];
      float p1 = __expf(v1 - Mrow[r]) * invS[r];
      c0 += p0; c1 += p1;
      pS[h * 512 + (r + halfOff) * 32 +      mcol] = f2bf(p0);
      pS[h * 512 + (r + halfOff) * 32 + 16 + mcol] = f2bf(p1);
    }
    c0 += __shfl_xor(c0, 16, 32);   // combine row halves -> full column sum
    c1 += __shfl_xor(c1, 16, 32);
    if (lane < 16) {
      unsafeAtomicAdd(&colFull[m0 +      lane], c0);
      unsafeAtomicAdd(&colFull[m0 + 16 + lane], c1);
    }

    // transpose P: C-layout -> A-layout via LDS bounce
    v16bf aP;
    #pragma unroll
    for (int r = 0; r < 8; ++r)
      #pragma unroll
      for (int e = 0; e < 2; ++e)
        aP[2 * r + e] = us2bf(pS[h * 512 + mcol * 32 + kidx(r, e, kadd)]);

    auto pvStep = [&](int t, v8f acc) -> v8f {
      v16bf bv;
      #pragma unroll
      for (int r = 0; r < 8; ++r)
        #pragma unroll
        for (int e = 0; e < 2; ++e) {
          int k = kidx(r, e, kadd);
          bv[2 * r + e] = us2bf(vS[k * DD + h * DH + t * 16 + mcol]);
        }
      return __builtin_amdgcn_wmma_f32_16x16x32_bf16(false, aP, false, bv, (short)0, acc, false, false);
    };
    O0 = pvStep(0, O0); O1 = pvStep(1, O1); O2 = pvStep(2, O2); O3 = pvStep(3, O3);
    __syncthreads();
  }

  // write O (already softmax-normalized, dead rows are exactly zero)
  #pragma unroll
  for (int r = 0; r < 8; ++r) {
    size_t base = ((size_t)(b * LL + l0 + r + halfOff)) * DD + h * DH + mcol;
    out[base +  0] = O0[r];
    out[base + 16] = O1[r];
    out[base + 32] = O2[r];
    out[base + 48] = O3[r];
  }

  // publish this block's column sums
  for (int i = tid; i < LL; i += 256)
    unsafeAtomicAdd(&wAcc[b * LL + i], colFull[i]);
}

__global__ __launch_bounds__(256)
void finalize_w_kernel(const float* __restrict__ wAcc, const float* __restrict__ doc,
                       float* __restrict__ wOut)
{
  __shared__ float red[256];
  const int b = blockIdx.x, tid = threadIdx.x;
  const float ds = doc[b];
  const int lenB = (int)ds;

  float x[4];
  #pragma unroll
  for (int kk = 0; kk < 4; ++kk) {
    int m = tid + kk * 256;
    float raw = wAcc[b * LL + m] * 0.125f / ds;   // mean over heads, /doc_size
    x[kk] = (m < lenB) ? raw : -3.0e38f;
  }
  float mx = fmaxf(fmaxf(x[0], x[1]), fmaxf(x[2], x[3]));
  red[tid] = mx; __syncthreads();
  for (int s = 128; s > 0; s >>= 1) { if (tid < s) red[tid] = fmaxf(red[tid], red[tid + s]); __syncthreads(); }
  mx = red[0]; __syncthreads();

  float e[4], ls = 0.0f;
  #pragma unroll
  for (int kk = 0; kk < 4; ++kk) {
    int m = tid + kk * 256;
    e[kk] = (m < lenB) ? __expf(x[kk] - mx) : 0.0f;
    ls += e[kk];
  }
  red[tid] = ls; __syncthreads();
  for (int s = 128; s > 0; s >>= 1) { if (tid < s) red[tid] += red[tid + s]; __syncthreads(); }
  float inv = (red[0] > 0.0f) ? (1.0f / red[0]) : 0.0f;
  #pragma unroll
  for (int kk = 0; kk < 4; ++kk) {
    int m = tid + kk * 256;
    wOut[b * LL + m] = e[kk] * inv;
  }
}

extern "C" void kernel_launch(void* const* d_in, const int* in_sizes, int n_in,
                              void* d_out, int out_size, void* d_ws, size_t ws_size,
                              hipStream_t stream) {
  (void)in_sizes; (void)n_in; (void)out_size; (void)ws_size;
  const float* Kg    = (const float*)d_in[0];
  const float* Qg    = (const float*)d_in[1];
  const float* Vg    = (const float*)d_in[2];
  const float* doc   = (const float*)d_in[3];
  const float* gamma = (const float*)d_in[4];
  const float* beta  = (const float*)d_in[5];
  float* out  = (float*)d_out;
  float* wOut = out + (size_t)4 * LL * DD;   // outputs concatenated: out then w
  float* wAcc = (float*)d_ws;

  const size_t NT = (size_t)4 * LL * DD;     // elements per tensor
  unsigned short* Kb = (unsigned short*)((char*)d_ws + 16384);
  unsigned short* Qb = Kb + NT;
  unsigned short* Vb = Qb + NT;

  hipMemsetAsync(d_ws, 0, 4 * LL * sizeof(float), stream);

  // pre-convert K/Q/V to bf16 so the Tensor Data Mover can stage tiles directly
  const int cvtBlocks = (int)(NT / 1024);
  to_bf16_kernel<<<dim3(cvtBlocks), dim3(256), 0, stream>>>(Kg, Kb);
  to_bf16_kernel<<<dim3(cvtBlocks), dim3(256), 0, stream>>>(Qg, Qb);
  to_bf16_kernel<<<dim3(cvtBlocks), dim3(256), 0, stream>>>(Vg, Vb);

  hipFuncSetAttribute((const void*)etc_attn_kernel,
                      hipFuncAttributeMaxDynamicSharedMemorySize, SMEM_BYTES);
  etc_attn_kernel<<<dim3(4 * (LL / TL)), dim3(256), SMEM_BYTES, stream>>>(
      Kb, Qb, Vb, doc, gamma, beta, out, wAcc);
  finalize_w_kernel<<<dim3(4), dim3(256), 0, stream>>>(wAcc, doc, wOut);
}
